// encoder_71614284694089
// MI455X (gfx1250) — matrix operand order repaired and checked by reading, not compiled, not run on hardware
//
#include <hip/hip_runtime.h>
#include <hip/hip_bf16.h>

// ---------------------------------------------------------------------------
// PointNet++ encoder for MI455X (gfx1250, wave32, WMMA).
//   B=8, N=8192, GEO=32, nsample=32 for all layers.
//   CFGS: (S, r, feat, mlp): (1024,0.1,3,[32,32,64]) (256,0.2,64,[64,64,128])
//                            (64,0.4,128,[128,128,256]) (16,0.8,256,[256,256,512])
// Weights are pre-packed into the native WMMA B-fragment layout so the hot
// loop is: 2x ds_load_b128 (A) + 2x global_load_b128 (B) + v_wmma per K-chunk.
// ---------------------------------------------------------------------------

typedef __attribute__((ext_vector_type(16))) _Float16 v16h;
typedef __attribute__((ext_vector_type(8)))  float    v8f;

#define BATCH 8
#define NPTS  8192
#define NS    32      // nsample
#define FPS_T 1024

// ---------------------------------------------------------------------------
// Farthest point sampling: one workgroup per batch, dist[] resident in LDS.
// Matches jax scan semantics: fidx[0]=0, then argmax of running min-dist.
// ---------------------------------------------------------------------------
__global__ void __launch_bounds__(FPS_T) fps_kernel(
    const float* __restrict__ xyz, int N, int S,
    int* __restrict__ fidx, float* __restrict__ nxyz)
{
    __shared__ float dist[NPTS];
    __shared__ float rv[FPS_T];
    __shared__ int   ri[FPS_T];

    const int b = blockIdx.x, tid = threadIdx.x;
    const float* base = xyz + (size_t)b * N * 3;

    for (int p = tid; p < N; p += FPS_T) dist[p] = 1e10f;
    __syncthreads();

    int far = 0;
    for (int j = 0; j < S; j++) {
        const float cx = base[far*3+0], cy = base[far*3+1], cz = base[far*3+2];
        if (tid == 0) {
            fidx[b*S + j] = far;
            nxyz[(b*S + j)*3 + 0] = cx;
            nxyz[(b*S + j)*3 + 1] = cy;
            nxyz[(b*S + j)*3 + 2] = cz;
        }
        float best = -1.0f; int bi = 0;
        for (int p = tid; p < N; p += FPS_T) {
            const float dx = base[p*3+0]-cx, dy = base[p*3+1]-cy, dz = base[p*3+2]-cz;
            const float d  = dx*dx + dy*dy + dz*dz;
            const float nd = fminf(dist[p], d);
            dist[p] = nd;
            if (nd > best) { best = nd; bi = p; }
        }
        rv[tid] = best; ri[tid] = bi;
        __syncthreads();
        for (int off = FPS_T/2; off > 0; off >>= 1) {
            if (tid < off) {
                const float ov = rv[tid+off]; const int oi = ri[tid+off];
                if (ov > rv[tid] || (ov == rv[tid] && oi < ri[tid])) { rv[tid] = ov; ri[tid] = oi; }
            }
            __syncthreads();
        }
        far = ri[0];
        __syncthreads();
    }
}

// ---------------------------------------------------------------------------
// Ball query: one wave32 per centroid. Scans points in index order and keeps
// the first 32 within radius (== reference's smallest-index selection),
// filling the remainder with the first hit.
// ---------------------------------------------------------------------------
__global__ void __launch_bounds__(128) ballq_kernel(
    const float* __restrict__ xyz, const float* __restrict__ nxyz,
    int N, int S, int B, float r2, int* __restrict__ oidx)
{
    __shared__ int sel[4][NS];
    const int wave = threadIdx.x >> 5, lane = threadIdx.x & 31;
    const int g = blockIdx.x * 4 + wave;
    if (g >= B * S) return;              // grids are multiples of 4 waves anyway
    const int b = g / S;
    const float cx = nxyz[g*3+0], cy = nxyz[g*3+1], cz = nxyz[g*3+2];
    const float* base = xyz + (size_t)b * N * 3;

    int cnt = 0;
    for (int p0 = 0; p0 < N && cnt < NS; p0 += 32) {
        const int p = p0 + lane;
        bool inr = false;
        if (p < N) {
            const float dx = base[p*3+0]-cx, dy = base[p*3+1]-cy, dz = base[p*3+2]-cz;
            inr = (dx*dx + dy*dy + dz*dz) <= r2;
        }
        const unsigned m = (unsigned)__ballot(inr);
        const int pos = cnt + __popc(m & ((1u << lane) - 1u));
        if (inr && pos < NS) sel[wave][pos] = p;
        cnt += __popc(m);
    }
    if (cnt == 0) { if (lane == 0) sel[wave][0] = 0; cnt = 1; }
    if (cnt > NS) cnt = NS;
    const int v = (lane < cnt) ? sel[wave][lane] : sel[wave][0];
    oidx[(size_t)g * NS + lane] = v;
}

// ---------------------------------------------------------------------------
// fp32 -> f16 weight pre-pack into the native WMMA B-fragment layout.
// B16 (32x16) fragment per ISA 7.12.2: lane = 16*((K>>3)&1 within chunk) + N%16,
// element i of the lane's v16h holds K = kc*32 + i + 8*(lane>>4) + ((i>=8)?8:0), N = col.
// Packed index: (((kc * (co/16)) + ntile) * 32 + lane) * 16 + i
// => each lane's fragment is one contiguous, 32B-aligned v16h.
// ---------------------------------------------------------------------------
__global__ void pack_w_kernel(const float* __restrict__ src,
                              _Float16* __restrict__ dst,
                              int Klog, int Kpad, int co)
{
    const int t = blockIdx.x * blockDim.x + threadIdx.x;
    if (t >= Kpad * co) return;
    const int k = t / co, n = t - k * co;
    const float v = (k < Klog) ? src[k * co + n] : 0.0f;
    const int kc = k >> 5, r = k & 31;
    const int lg = (r >> 3) & 1, i = (r & 7) | ((r >> 4) << 3);
    const int ntile = n >> 4, lr = n & 15;
    const int lane = (lg << 4) | lr;
    dst[(((size_t)kc * (co >> 4) + ntile) * 32 + lane) * 16 + i] = (_Float16)v;
}

// ---------------------------------------------------------------------------
// One MLP layer via v_wmma_f32_16x16x32_f16; K, CO compile-time constants.
//   src: LDS f16 [32 x K].  Wp: packed B fragments in global.  dst: LDS.
// 4 waves split the 2 x (CO/16) output tiles; loops are wave-uniform so EXEC
// stays all-ones around the WMMA (ISA requirement).
// Fragment layouts per CDNA5 ISA 7.12.2:
//   A16 (16x32): M = lane&15, K = i + 8*(lane>>4) + ((i>=8)?8:0)
//   C32 (16x16): M = r + 8*(lane>>4),  N = lane&15
// ---------------------------------------------------------------------------
template<int K, int CO>
__device__ __forceinline__ void mlp_layer(
    const _Float16* __restrict__ src,
    const v16h* __restrict__ Wp, const float* __restrict__ bias,
    _Float16* __restrict__ dst, int tid)
{
    const int wave = tid >> 5, lane = tid & 31;
    const int lg = lane >> 4, lr = lane & 15;
    constexpr int NT = CO >> 4;
    constexpr int KC = K >> 5;
    for (int t = wave; t < NT * 2; t += 4) {
        const int mtile = t & 1, ntile = t >> 1;
        const int n = (ntile << 4) + lr;
        const int m = (mtile << 4) + lr;
        const _Float16* arow = src + m * K + (lg << 3);
        const v16h* wrow = Wp + (size_t)ntile * 32 + lane;
        v8f acc = {};
#pragma unroll
        for (int kc = 0; kc < KC; kc++) {
            v16h a;
#pragma unroll
            for (int i = 0; i < 16; i++)      // -> 2x ds_load_b128
                a[i] = arow[kc*32 + i + ((i & 8) ? 8 : 0)];
            const v16h bf = wrow[(size_t)kc * NT * 32];  // -> 2x global_load_b128
            acc = __builtin_amdgcn_wmma_f32_16x16x32_f16(
                false, a, false, bf, (short)0, acc, false, false);
        }
        const float bv = bias[n];
#pragma unroll
        for (int r = 0; r < 8; r++) {
            float v = fmaxf(acc[r] + bv, 0.0f);          // fused ReLU
            dst[((mtile << 4) + r + (lg << 3)) * CO + n] = (_Float16)v;
        }
    }
}

// ---------------------------------------------------------------------------
// Set-abstraction layer body: one workgroup (128 thr = 4 waves) per centroid.
// Gather+geo-embed -> 3 WMMA MLP layers (LDS ping-pong) -> max over nsample.
// All dims are template constants -> fully unrolled K loops, static LDS.
// ---------------------------------------------------------------------------
template<int CPTS, int K1, int CO1, int CO2, int CO3>
__global__ void __launch_bounds__(128) sa_kernel(
    const float* __restrict__ xyz_in, const float* __restrict__ pts_in,
    int Nin, int S,
    const float* __restrict__ nxyz, const int* __restrict__ ballidx,
    const float* __restrict__ geo_w, const float* __restrict__ geo_b,
    const v16h* __restrict__ W1, const float* __restrict__ b1,
    const v16h* __restrict__ W2, const float* __restrict__ b2,
    const v16h* __restrict__ W3, const float* __restrict__ b3,
    float* __restrict__ pts_out)
{
    constexpr int SZA = 32 * (K1 > CO2 ? K1 : CO2);
    constexpr int SZB = 32 * (CO1 > CO3 ? CO1 : CO3);
    __shared__ _Float16 bufA[SZA];   // input tile / layer-2 out
    __shared__ _Float16 bufB[SZB];   // layer-1 out / layer-3 out
    __shared__ float gx[NS], gy[NS], gz[NS];
    __shared__ int   li[NS];

    const int tid = threadIdx.x;
    const int g = blockIdx.x;            // b*S + s
    const int b = g / S;
    const float cx = nxyz[g*3+0], cy = nxyz[g*3+1], cz = nxyz[g*3+2];

    // Warm L2/WGP$ for the packed weights while we gather (global_prefetch_b8).
    __builtin_prefetch((const char*)W1 + (size_t)tid * 128, 0, 0);
    __builtin_prefetch((const char*)W3 + (size_t)tid * 128, 0, 0);

    if (tid < NS) {
        const int i = ballidx[(size_t)g * NS + tid];
        li[tid] = i;
        const float* p = xyz_in + (size_t)(b * Nin + i) * 3;
        gx[tid] = p[0] - cx; gy[tid] = p[1] - cy; gz[tid] = p[2] - cz;
    }
    __syncthreads();

    // Build 32 x K1 activation tile: [ relu(g_xyz @ geo_w + geo_b) | pts | 0-pad ]
    for (int pos = tid; pos < NS * K1; pos += 128) {
        const int r = pos / K1, c = pos - r * K1;
        float v = 0.0f;
        if (c < 32) {
            v = fmaxf(gx[r]*geo_w[c] + gy[r]*geo_w[32+c] + gz[r]*geo_w[64+c] + geo_b[c], 0.0f);
        } else if (c < 32 + CPTS) {
            v = pts_in[(size_t)(b * Nin + li[r]) * CPTS + (c - 32)];
        }
        bufA[pos] = (_Float16)v;
    }
    __syncthreads();

    mlp_layer<K1,  CO1>(bufA, W1, b1, bufB, tid);  __syncthreads();
    mlp_layer<CO1, CO2>(bufB, W2, b2, bufA, tid);  __syncthreads();
    mlp_layer<CO2, CO3>(bufA, W3, b3, bufB, tid);  __syncthreads();

    // Max-pool over the 32 neighbors.
    for (int c = tid; c < CO3; c += 128) {
        float mx = -3.4e38f;
#pragma unroll 8
        for (int r = 0; r < NS; r++) mx = fmaxf(mx, (float)bufB[r * CO3 + c]);
        pts_out[(size_t)g * CO3 + c] = mx;
    }
}

// ---------------------------------------------------------------------------
// Final: out[b,0,c] = sum_s pts[b,s,c] * w[0,s] + bias[0]    (8 x 512)
// ---------------------------------------------------------------------------
__global__ void final_conv_kernel(const float* __restrict__ pts,
                                  const float* __restrict__ w,
                                  const float* __restrict__ bias,
                                  float* __restrict__ out)
{
    const int t = blockIdx.x * blockDim.x + threadIdx.x;
    if (t >= BATCH * 512) return;
    const int b = t / 512, c = t - b * 512;
    float acc = bias[0];
#pragma unroll
    for (int s = 0; s < 16; s++) acc += pts[(size_t)(b*16 + s)*512 + c] * w[s];
    out[t] = acc;
}

// ---------------------------------------------------------------------------
extern "C" void kernel_launch(void* const* d_in, const int* in_sizes, int n_in,
                              void* d_out, int out_size, void* d_ws, size_t ws_size,
                              hipStream_t stream) {
    (void)in_sizes; (void)n_in; (void)out_size; (void)ws_size;

    const float* xyz = (const float*)d_in[0];
    const float* sap[4][8];
    for (int l = 0; l < 4; l++)
        for (int j = 0; j < 8; j++)
            sap[l][j] = (const float*)d_in[1 + l*8 + j];
    const float* convw = (const float*)d_in[33];
    const float* convb = (const float*)d_in[34];

    static const int   Sarr[4]    = {1024, 256, 64, 16};
    static const float rad2[4]    = {0.01f, 0.04f, 0.16f, 0.64f};
    static const int   coArr[4][3]= {{32,32,64},{64,64,128},{128,128,256},{256,256,512}};
    static const int   K1pad[4]   = {64, 96, 160, 288};  // cin1 = 32+feat, padded to %32
    static const int   K1log[4]   = {35, 96, 160, 288};

    // Workspace carve-out (all buffers fully written before being read).
    char* ws = (char*)d_ws; size_t off = 0;
    auto alloc = [&](size_t bytes) -> void* {
        void* p = ws + off; off += (bytes + 255) & ~(size_t)255; return p;
    };
    int*   fidx = (int*)alloc((size_t)BATCH * 1024 * sizeof(int));
    float* nx[4];
    for (int l = 0; l < 4; l++) nx[l] = (float*)alloc((size_t)BATCH * Sarr[l] * 3 * sizeof(float));
    int*   bidx = (int*)alloc((size_t)BATCH * 1024 * NS * sizeof(int));
    float* pts[4];
    for (int l = 0; l < 4; l++) pts[l] = (float*)alloc((size_t)BATCH * Sarr[l] * coArr[l][2] * sizeof(float));

    const v16h* Wh[4][3];
    for (int l = 0; l < 4; l++) {
        const int kpad[3] = {K1pad[l], coArr[l][0], coArr[l][1]};
        const int klog[3] = {K1log[l], coArr[l][0], coArr[l][1]};
        for (int j = 0; j < 3; j++) {
            const int ncol = coArr[l][j];
            _Float16* dst = (_Float16*)alloc((size_t)kpad[j] * ncol * sizeof(_Float16));
            Wh[l][j] = (const v16h*)dst;
            const int tot = kpad[j] * ncol;
            pack_w_kernel<<<(tot + 255)/256, 256, 0, stream>>>(
                sap[l][2 + 2*j], dst, klog[j], kpad[j], ncol);
        }
    }

    for (int l = 0; l < 4; l++) {
        const int Nin  = (l == 0) ? NPTS : Sarr[l-1];
        const float* xin = (l == 0) ? xyz : nx[l-1];
        const float* pin = (l == 0) ? xyz : pts[l-1];
        const int S = Sarr[l];

        fps_kernel<<<BATCH, FPS_T, 0, stream>>>(xin, Nin, S, fidx, nx[l]);

        const int nwaves = BATCH * S;
        ballq_kernel<<<(nwaves + 3)/4, 128, 0, stream>>>(xin, nx[l], Nin, S, BATCH, rad2[l], bidx);

        switch (l) {
        case 0:
            sa_kernel<3, 64, 32, 32, 64><<<BATCH*S, 128, 0, stream>>>(
                xin, pin, Nin, S, nx[l], bidx, sap[l][0], sap[l][1],
                Wh[l][0], sap[l][3], Wh[l][1], sap[l][5], Wh[l][2], sap[l][7], pts[l]);
            break;
        case 1:
            sa_kernel<64, 96, 64, 64, 128><<<BATCH*S, 128, 0, stream>>>(
                xin, pin, Nin, S, nx[l], bidx, sap[l][0], sap[l][1],
                Wh[l][0], sap[l][3], Wh[l][1], sap[l][5], Wh[l][2], sap[l][7], pts[l]);
            break;
        case 2:
            sa_kernel<128, 160, 128, 128, 256><<<BATCH*S, 128, 0, stream>>>(
                xin, pin, Nin, S, nx[l], bidx, sap[l][0], sap[l][1],
                Wh[l][0], sap[l][3], Wh[l][1], sap[l][5], Wh[l][2], sap[l][7], pts[l]);
            break;
        default:
            sa_kernel<256, 288, 256, 256, 512><<<BATCH*S, 128, 0, stream>>>(
                xin, pin, Nin, S, nx[l], bidx, sap[l][0], sap[l][1],
                Wh[l][0], sap[l][3], Wh[l][1], sap[l][5], Wh[l][2], sap[l][7], pts[l]);
            break;
        }
    }

    final_conv_kernel<<<(BATCH*512 + 255)/256, 256, 0, stream>>>(pts[3], convw, convb, (float*)d_out);
}
